// LSTMCell_74938589381014
// MI455X (gfx1250) — compile-verified, matrix operand order checked by reading, not run on hardware
//
#include <hip/hip_runtime.h>

#if defined(__has_builtin)
#  if __has_builtin(__builtin_amdgcn_global_load_async_to_lds_b128)
#    define USE_ASYNC 1
#  endif
#endif

typedef __attribute__((ext_vector_type(16))) __bf16 v16bf;
typedef __attribute__((ext_vector_type(8)))  float  v8f;

#ifdef USE_ASYNC
typedef int v4i __attribute__((vector_size(16)));
typedef __attribute__((address_space(1))) v4i* gvec_p;   // pointer to AS1 int4
typedef __attribute__((address_space(3))) v4i* lvec_p;   // pointer to AS3 int4
#endif

#define BATCH   8192
#define INDIM   512
#define HID     1024
#define DCAT    1536   // INDIM + HID

#define BM 128
#define BN 64
#define BK 64          // two WMMA K-steps per LDS stage

// ---------- helpers ----------

__device__ __forceinline__ unsigned short f2bf(float f) {
    unsigned int u = __float_as_uint(f);
    u += 0x7FFFu + ((u >> 16) & 1u);           // round-to-nearest-even
    return (unsigned short)(u >> 16);
}

__device__ __forceinline__ unsigned int pack2(float a, float b) {
    return (unsigned int)f2bf(a) | ((unsigned int)f2bf(b) << 16);
}

// sigmoid via v_exp_f32 + v_rcp_f32 (compact, no branches)
__device__ __forceinline__ float fast_sig(float x) {
    return __builtin_amdgcn_rcpf(1.0f + __expf(-x));
}
__device__ __forceinline__ float fast_tanh(float x) {
    return 2.0f * fast_sig(2.0f * x) - 1.0f;
}

union FragU {
    v16bf v;
    uint4 u[2];
};

// 16x32 bf16 fragment (WMMA A-matrix lane layout) from LDS tile with row
// stride 64 bf16. Lane l(0..15): row base+l, K=kk+0..7 then kk+16..23;
// lane 16+l: K=kk+8..15 then kk+24..31. Two ds_load_b128.
__device__ __forceinline__ v16bf load_frag64(const unsigned short* lds, int rowBase,
                                             int kk, int lane) {
    int r    = rowBase + (lane & 15);
    int kofs = kk + ((lane >> 4) << 3);
    FragU f;
    f.u[0] = *(const uint4*)(lds + r * 64 + kofs);
    f.u[1] = *(const uint4*)(lds + r * 64 + kofs + 16);
    return f.v;
}

// Copy ROWS x 64 bf16 tile (row stride ld elems) global -> LDS.
// Async path: GLOBAL_LOAD_ASYNC_TO_LDS_B128 (16B per lane, ASYNCcnt).
template <int ROWS>
__device__ __forceinline__ void tile_copy(const unsigned short* __restrict__ g,
                                          int ld, unsigned short* s, int tid)
{
#pragma unroll
    for (int i = 0; i < (ROWS * 8) / 256; ++i) {
        int idx = tid + i * 256;
        int r = idx >> 3, c = idx & 7;
#ifdef USE_ASYNC
        __builtin_amdgcn_global_load_async_to_lds_b128(
            (gvec_p)(g + (size_t)r * ld + c * 8),
            (lvec_p)(s + r * 64 + c * 8),
            0, 0);
#else
        uint4 d = *(const uint4*)(g + (size_t)r * ld + c * 8);
        *(uint4*)(s + r * 64 + c * 8) = d;
#endif
    }
}

__device__ __forceinline__ void wait_async() {
#ifdef USE_ASYNC
#  if __has_builtin(__builtin_amdgcn_s_wait_asynccnt)
    __builtin_amdgcn_s_wait_asynccnt(0);
#  else
    asm volatile("s_wait_asynccnt 0" ::: "memory");
#  endif
#endif
}

// ---------- conversion kernels ----------

// v = bf16(cat(x, h)) : [BATCH, DCAT]
__global__ __launch_bounds__(256) void concat_cvt_kernel(
    const float* __restrict__ x, const float* __restrict__ h,
    unsigned int* __restrict__ v)
{
    int t = blockIdx.x * blockDim.x + threadIdx.x;      // pair index
    int b = t / (DCAT / 2);
    int d = (t - b * (DCAT / 2)) * 2;
    float f0, f1;
    if (d < INDIM) {
        f0 = x[(size_t)b * INDIM + d];
        f1 = x[(size_t)b * INDIM + d + 1];
    } else {
        f0 = h[(size_t)b * HID + (d - INDIM)];
        f1 = h[(size_t)b * HID + (d - INDIM) + 1];
    }
    v[t] = pack2(f0, f1);
}

// Wt[g][n][k] = bf16(W[g][k][n]); K = Kdim (1536 for W1, 1024 for W2).
// One-time transpose so GEMM B-tiles are contiguous rows (async-copy friendly).
__global__ __launch_bounds__(256) void cvtT_kernel(
    const float* __restrict__ in, unsigned int* __restrict__ out, int Kdim)
{
    int t  = blockIdx.x * blockDim.x + threadIdx.x;     // output pair index
    int KP = Kdim >> 1;
    int g  = t / (HID * KP);
    int r  = t - g * HID * KP;
    int n  = r / KP;
    int k  = (r - n * KP) * 2;
    const float* p = in + ((size_t)g * Kdim + k) * HID + n;
    out[t] = pack2(p[0], p[HID]);
}

// ---------- GEMM1: hdn[g] = bf16(tanh(v @ W1[g] + b1[g])) ----------

__global__ __launch_bounds__(256) void gemm1_tanh_kernel(
    const unsigned short* __restrict__ v,     // [BATCH, DCAT] bf16
    const unsigned short* __restrict__ w1t,   // [4, HID, DCAT] bf16 (transposed)
    const float* __restrict__ b1,             // [4, HID]
    unsigned short* __restrict__ hdn)         // [4, BATCH, HID] bf16
{
    __shared__ __align__(16) unsigned short sA[2][BM * BK];   // [m][k]
    __shared__ __align__(16) unsigned short sB[2][BN * BK];   // [n][k]

    const int tid  = threadIdx.x;
    const int wave = tid >> 5;
    const int lane = tid & 31;
    const int wm   = (wave & 3) * 32;
    const int wn   = (wave >> 2) * 32;
    const int m0   = blockIdx.y * BM;
    const int n0   = blockIdx.x * BN;
    const int g    = blockIdx.z;

    const unsigned short* Aptr = v   + (size_t)m0 * DCAT;
    const unsigned short* Bptr = w1t + ((size_t)g * HID + n0) * DCAT;
    unsigned short*       Hg   = hdn + (size_t)g * BATCH * HID;

    v8f acc[2][2] = {};

    constexpr int S = DCAT / BK;                // 24 stages
    tile_copy<BM>(Aptr, DCAT, sA[0], tid);
    tile_copy<BN>(Bptr, DCAT, sB[0], tid);
    wait_async();
    __syncthreads();

    for (int s = 0; s < S; ++s) {
        int cur = s & 1, nxt = cur ^ 1;
        if (s + 1 < S) {                        // prefetch next stage (async DMA)
            tile_copy<BM>(Aptr + (s + 1) * BK, DCAT, sA[nxt], tid);
            tile_copy<BN>(Bptr + (s + 1) * BK, DCAT, sB[nxt], tid);
        }
#pragma unroll
        for (int ks = 0; ks < BK; ks += 32) {
            v16bf a0 = load_frag64(sA[cur], wm,      ks, lane);
            v16bf a1 = load_frag64(sA[cur], wm + 16, ks, lane);
            v16bf b0 = load_frag64(sB[cur], wn,      ks, lane);
            v16bf b1f= load_frag64(sB[cur], wn + 16, ks, lane);
            acc[0][0] = __builtin_amdgcn_wmma_f32_16x16x32_bf16(false, a0, false, b0,  (short)0, acc[0][0], false, false);
            acc[0][1] = __builtin_amdgcn_wmma_f32_16x16x32_bf16(false, a0, false, b1f, (short)0, acc[0][1], false, false);
            acc[1][0] = __builtin_amdgcn_wmma_f32_16x16x32_bf16(false, a1, false, b0,  (short)0, acc[1][0], false, false);
            acc[1][1] = __builtin_amdgcn_wmma_f32_16x16x32_bf16(false, a1, false, b1f, (short)0, acc[1][1], false, false);
        }
        wait_async();
        __syncthreads();
    }

    const float* bg = b1 + (size_t)g * HID;
#pragma unroll
    for (int mt = 0; mt < 2; ++mt) {
#pragma unroll
        for (int nt = 0; nt < 2; ++nt) {
            int mbase = m0 + wm + mt * 16 + ((lane >> 4) << 3);
            int n     = n0 + wn + nt * 16 + (lane & 15);
            float bias = bg[n];
#pragma unroll
            for (int i = 0; i < 8; ++i) {
                float t = fast_tanh(acc[mt][nt][i] + bias);
                Hg[(size_t)(mbase + i) * HID + n] = f2bf(t);
            }
        }
    }
}

// ---------- GEMM2 (4 gates per block) + fused LSTM epilogue ----------

__global__ __launch_bounds__(256) void gemm2_lstm_kernel(
    const unsigned short* __restrict__ hdn,   // [4, BATCH, HID] bf16
    const unsigned short* __restrict__ w2t,   // [4, HID, HID] bf16 (transposed)
    const float* __restrict__ b2,             // [4, HID]
    const float* __restrict__ c_t,            // [BATCH, HID]
    float* __restrict__ out)                  // [2, BATCH, HID]: c_i then output
{
    __shared__ __align__(16) unsigned short sA[2][BM * BK];
    __shared__ __align__(16) unsigned short sB[2][BN * BK];

    const int tid  = threadIdx.x;
    const int wave = tid >> 5;
    const int lane = tid & 31;
    const int wm   = (wave & 3) * 32;
    const int wn   = (wave >> 2) * 32;
    const int m0   = blockIdx.y * BM;
    const int n0   = blockIdx.x * BN;

    v8f acc[4][2][2] = {};

    constexpr int S = HID / BK;                 // 16 stages per gate
    for (int g = 0; g < 4; ++g) {
        const unsigned short* Aptr = hdn + (size_t)g * BATCH * HID + (size_t)m0 * HID;
        const unsigned short* Bptr = w2t + ((size_t)g * HID + n0) * HID;

        tile_copy<BM>(Aptr, HID, sA[0], tid);
        tile_copy<BN>(Bptr, HID, sB[0], tid);
        wait_async();
        __syncthreads();

        for (int s = 0; s < S; ++s) {
            int cur = s & 1, nxt = cur ^ 1;
            if (s + 1 < S) {
                tile_copy<BM>(Aptr + (s + 1) * BK, HID, sA[nxt], tid);
                tile_copy<BN>(Bptr + (s + 1) * BK, HID, sB[nxt], tid);
            }
#pragma unroll
            for (int ks = 0; ks < BK; ks += 32) {
                v16bf a0 = load_frag64(sA[cur], wm,      ks, lane);
                v16bf a1 = load_frag64(sA[cur], wm + 16, ks, lane);
                v16bf b0 = load_frag64(sB[cur], wn,      ks, lane);
                v16bf b1f= load_frag64(sB[cur], wn + 16, ks, lane);
                acc[g][0][0] = __builtin_amdgcn_wmma_f32_16x16x32_bf16(false, a0, false, b0,  (short)0, acc[g][0][0], false, false);
                acc[g][0][1] = __builtin_amdgcn_wmma_f32_16x16x32_bf16(false, a0, false, b1f, (short)0, acc[g][0][1], false, false);
                acc[g][1][0] = __builtin_amdgcn_wmma_f32_16x16x32_bf16(false, a1, false, b0,  (short)0, acc[g][1][0], false, false);
                acc[g][1][1] = __builtin_amdgcn_wmma_f32_16x16x32_bf16(false, a1, false, b1f, (short)0, acc[g][1][1], false, false);
            }
            wait_async();
            __syncthreads();
        }
    }

    const size_t BH = (size_t)BATCH * HID;
#pragma unroll
    for (int mt = 0; mt < 2; ++mt) {
#pragma unroll
        for (int nt = 0; nt < 2; ++nt) {
            int mbase = m0 + wm + mt * 16 + ((lane >> 4) << 3);
            int n     = n0 + wn + nt * 16 + (lane & 15);
            float bi = b2[0 * HID + n];
            float bf = b2[1 * HID + n];
            float bo = b2[2 * HID + n];
            float bz = b2[3 * HID + n];
#pragma unroll
            for (int i = 0; i < 8; ++i) {
                int m = mbase + i;
                float vi = acc[0][mt][nt][i] + bi;
                float vf = acc[1][mt][nt][i] + bf;
                float vo = acc[2][mt][nt][i] + bo;
                float vz = acc[3][mt][nt][i] + bz;
                float ct = c_t[(size_t)m * HID + n];
                float ci = fast_sig(vf) * ct + fast_sig(vi) * fast_tanh(vz);
                out[(size_t)m * HID + n]      = ci;
                out[BH + (size_t)m * HID + n] = fast_sig(vo);
            }
        }
    }
}

// ---------- launcher ----------

extern "C" void kernel_launch(void* const* d_in, const int* in_sizes, int n_in,
                              void* d_out, int out_size, void* d_ws, size_t ws_size,
                              hipStream_t stream) {
    const float* x   = (const float*)d_in[0];   // [8192, 512]
    const float* h_t = (const float*)d_in[1];   // [8192, 1024]
    const float* c_t = (const float*)d_in[2];   // [8192, 1024]
    const float* W1  = (const float*)d_in[3];   // [4, 1536, 1024]
    const float* b1  = (const float*)d_in[4];   // [4, 1024]
    const float* W2  = (const float*)d_in[5];   // [4, 1024, 1024]
    const float* b2  = (const float*)d_in[6];   // [4, 1024]
    float* out = (float*)d_out;

    // workspace layout (bf16 copies; weights stored transposed)
    char* ws = (char*)d_ws;
    const size_t SZ_V   = (size_t)BATCH * DCAT * 2;          // 25,165,824
    const size_t SZ_W1  = (size_t)4 * DCAT * HID * 2;        // 12,582,912
    const size_t SZ_W2  = (size_t)4 * HID * HID * 2;         //  8,388,608
    unsigned short* v_bf    = (unsigned short*)(ws);
    unsigned short* w1t_bf  = (unsigned short*)(ws + SZ_V);
    unsigned short* w2t_bf  = (unsigned short*)(ws + SZ_V + SZ_W1);
    unsigned short* hdn_bf  = (unsigned short*)(ws + SZ_V + SZ_W1 + SZ_W2);

    // 1) convert / concat / transpose to bf16
    {
        int pairs = BATCH * DCAT / 2;                        // 6,291,456
        concat_cvt_kernel<<<pairs / 256, 256, 0, stream>>>(x, h_t, (unsigned int*)v_bf);
    }
    {
        int pairs = 4 * DCAT * HID / 2;                      // 3,145,728
        cvtT_kernel<<<pairs / 256, 256, 0, stream>>>(W1, (unsigned int*)w1t_bf, DCAT);
    }
    {
        int pairs = 4 * HID * HID / 2;                       // 2,097,152
        cvtT_kernel<<<pairs / 256, 256, 0, stream>>>(W2, (unsigned int*)w2t_bf, HID);
    }

    // 2) GEMM1 + tanh -> hdn (bf16)
    {
        dim3 grid(HID / BN, BATCH / BM, 4);                  // (16, 64, 4)
        gemm1_tanh_kernel<<<grid, 256, 0, stream>>>(v_bf, w1t_bf, b1, hdn_bf);
    }

    // 3) GEMM2 (all 4 gates per block) + fused LSTM epilogue
    {
        dim3 grid(HID / BN, BATCH / BM);                     // (16, 64)
        gemm2_lstm_kernel<<<grid, 256, 0, stream>>>(hdn_bf, w2t_bf, b2, c_t, out);
    }
}
